// MultiHeadAttention_74517682585786
// MI455X (gfx1250) — matrix operand order repaired and checked by reading, not compiled, not run on hardware
//
#include <hip/hip_runtime.h>

typedef __bf16 bf16;
typedef __attribute__((ext_vector_type(16))) __bf16 v16bf;
typedef __attribute__((ext_vector_type(8)))  __bf16 v8bf;
typedef __attribute__((ext_vector_type(4)))  __bf16 v4bf;
typedef __attribute__((ext_vector_type(2)))  __bf16 v2bf;
typedef __attribute__((ext_vector_type(8)))  float  v8f;
typedef __attribute__((ext_vector_type(4)))  int    v4i;

#define EMB   1024
#define HEADS 16
#define HDIM  64
#define SEQ   2048
#define BATCH 4
#define MTOT  (BATCH * SEQ)   // 8192 rows

#if __has_builtin(__builtin_amdgcn_global_load_async_to_lds_b128) && \
    __has_builtin(__builtin_amdgcn_s_wait_asynccnt)
#define HAVE_ASYNC_LDS 1
#else
#define HAVE_ASYNC_LDS 0
#endif

// Pointer types matching the async builtin's parameters (probe-verified):
//   arg0: v4i* in global addrspace, arg1: v4i* in LDS addrspace
typedef __attribute__((address_space(1))) v4i* gptr_b128;
typedef __attribute__((address_space(3))) v4i* lptr_b128;

static __device__ __forceinline__ v8f wmma_bf16(v16bf a, v16bf b, v8f c) {
    // D = A(16x32 bf16) * B(32x16 bf16) + C(16x16 f32)
    return __builtin_amdgcn_wmma_f32_16x16x32_bf16(
        false, a, false, b, (short)0, c, false, false);
}

// 16-lane reductions entirely in VALU via DPP16 (no ds_bpermute round-trips).
#define DPP_STEP_MAX(x, ctrl)                                                  \
    x = fmaxf(x, __int_as_float(__builtin_amdgcn_mov_dpp(                      \
                     __float_as_int(x), (ctrl), 0xF, 0xF, true)))
#define DPP_STEP_ADD(x, ctrl)                                                  \
    x = x + __int_as_float(__builtin_amdgcn_mov_dpp(                           \
                __float_as_int(x), (ctrl), 0xF, 0xF, true))
#define RED16_MAX(x)                                                           \
    do { DPP_STEP_MAX(x, 0xB1); DPP_STEP_MAX(x, 0x4E);                         \
         DPP_STEP_MAX(x, 0x141); DPP_STEP_MAX(x, 0x140); } while (0)
#define RED16_ADD(x)                                                           \
    do { DPP_STEP_ADD(x, 0xB1); DPP_STEP_ADD(x, 0x4E);                         \
         DPP_STEP_ADD(x, 0x141); DPP_STEP_ADD(x, 0x140); } while (0)

#define CONCAT16(lo, hi)                                                       \
    __builtin_shufflevector(lo, hi, 0, 1, 2, 3, 4, 5, 6, 7, 8, 9, 10, 11, 12, \
                            13, 14, 15)
#define DEINTERLEAVE16(w)                                                      \
    __builtin_shufflevector(w, w, 0, 2, 4, 6, 8, 10, 12, 14, 1, 3, 5, 7, 9,   \
                            11, 13, 15)

// ---------------------------------------------------------------------------
// f32 -> bf16 conversion (vectorized 4-wide)
// ---------------------------------------------------------------------------
__global__ __launch_bounds__(256) void cvt_f32_to_bf16(
    const float* __restrict__ in, bf16* __restrict__ out, int n4) {
    int i = blockIdx.x * blockDim.x + threadIdx.x;
    if (i >= n4) return;
    float4 f = ((const float4*)in)[i];
    v4bf o;
    o[0] = (bf16)f.x; o[1] = (bf16)f.y; o[2] = (bf16)f.z; o[3] = (bf16)f.w;
    ((v4bf*)out)[i] = o;
}

// ---------------------------------------------------------------------------
// GEMM: C[M,N] = A[M,K](bf16) * W[N,K]^T(bf16) + bias.
// Block = 8 waves, 256 rows x 64 cols; each wave owns a 32x64 tile (two A
// fragments reuse every B fragment in registers -> 8 WMMA per staged chunk).
// The 64-col B strip is shared block-wide and staged (double-buffered) into
// LDS with GLOBAL_LOAD_ASYNC_TO_LDS_B128 when available.
// MODE 0: f32 row-major [M,N]           (final output projection)
// MODE 1: bf16 scatter to [B,H,SEQ,D]   (Q / K), value scaled by ascale
// MODE 2: bf16 scatter to [B,H,D,SEQ]   (V transposed)
// ---------------------------------------------------------------------------
template <int MODE>
__global__ __launch_bounds__(256) void gemm_xWT(
    const bf16* __restrict__ A, const bf16* __restrict__ W,
    const float* __restrict__ bias, void* __restrict__ out,
    int M, int K, int N, float ascale) {
    const int wid  = threadIdx.x >> 5;
    const int lane = threadIdx.x & 31;
    const int g = lane >> 4;       // half-wave group
    const int l = lane & 15;
    const int nblk = N >> 6;
    const int bm = blockIdx.x / nblk, bn = blockIdx.x % nblk;
    const int m0 = (bm * 8 + wid) << 5;    // 32 rows per wave
    const int n0 = bn << 6;

    v8f acc[2][4] = {};
    const bf16* arow0 = A + (size_t)(m0 + l) * K + g * 8;   // A-frag, rows m0..
    const bf16* arow1 = arow0 + (size_t)16 * K;             // A-frag, rows m0+16..

#if HAVE_ASYNC_LDS
    // B strip k-chunk: 64 cols x 32 k = 4KB, double buffered.
    __shared__ __align__(64) bf16 sb[2][64 * 32];
    const int scol = threadIdx.x >> 2, schunk = threadIdx.x & 3;
    auto gW = (const __attribute__((address_space(1))) bf16*)W;
    auto sW = (__attribute__((address_space(3))) bf16*)&sb[0][0];
    const __attribute__((address_space(1))) bf16* gsrc =
        gW + (size_t)(n0 + scol) * K + schunk * 8;
    __attribute__((address_space(3))) bf16* sdst = sW + scol * 32 + schunk * 8;

    __builtin_amdgcn_global_load_async_to_lds_b128((gptr_b128)gsrc,
                                                   (lptr_b128)sdst, 0, 0);
    int cur = 0;
    for (int k0 = 0; k0 < K; k0 += 32) {
        __builtin_amdgcn_s_wait_asynccnt(0);   // this wave's stage done
        __syncthreads();                       // all waves' stage visible
        if (k0 + 32 < K)
            __builtin_amdgcn_global_load_async_to_lds_b128(
                (gptr_b128)(gsrc + k0 + 32),
                (lptr_b128)(sdst + (cur ^ 1) * (64 * 32)), 0, 0);

        v16bf a0 = CONCAT16(*(const v8bf*)(arow0 + k0),
                            *(const v8bf*)(arow0 + k0 + 16));
        v16bf a1 = CONCAT16(*(const v8bf*)(arow1 + k0),
                            *(const v8bf*)(arow1 + k0 + 16));
#pragma unroll
        for (int j = 0; j < 4; ++j) {
            v16bf b = CONCAT16(*(const v8bf*)&sb[cur][(j * 16 + l) * 32 + g * 8],
                               *(const v8bf*)&sb[cur][(j * 16 + l) * 32 + g * 8 + 16]);
            acc[0][j] = wmma_bf16(a0, b, acc[0][j]);
            acc[1][j] = wmma_bf16(a1, b, acc[1][j]);
        }
        __syncthreads();                       // all waves done reading cur
        cur ^= 1;
    }
#else
    const bf16* wrow = W + (size_t)(n0 + l) * K + g * 8;   // B-frag lane base
    for (int k0 = 0; k0 < K; k0 += 32) {
        v16bf a0 = CONCAT16(*(const v8bf*)(arow0 + k0),
                            *(const v8bf*)(arow0 + k0 + 16));
        v16bf a1 = CONCAT16(*(const v8bf*)(arow1 + k0),
                            *(const v8bf*)(arow1 + k0 + 16));
#pragma unroll
        for (int j = 0; j < 4; ++j) {
            v16bf b = CONCAT16(*(const v8bf*)(wrow + (size_t)j * 16 * K + k0),
                               *(const v8bf*)(wrow + (size_t)j * 16 * K + k0 + 16));
            acc[0][j] = wmma_bf16(a0, b, acc[0][j]);
            acc[1][j] = wmma_bf16(a1, b, acc[1][j]);
        }
    }
#endif

#pragma unroll
    for (int mi = 0; mi < 2; ++mi) {
#pragma unroll
        for (int j = 0; j < 4; ++j) {
            const int col = n0 + j * 16 + l;
            const float bv = bias[col];
#pragma unroll
            for (int r = 0; r < 8; ++r) {
                const int row = m0 + mi * 16 + r + 8 * g;  // C: VGPR r -> M=r+8g
                const float v = (acc[mi][j][r] + bv) * ascale;
                if (MODE == 0) {
                    ((float*)out)[(size_t)row * N + col] = v;
                } else {
                    const int b_ = row >> 11, n_ = row & (SEQ - 1);
                    const int h_ = col >> 6, d_ = col & (HDIM - 1);
                    if (MODE == 1)
                        ((bf16*)out)[((size_t)((b_ * HEADS + h_) * SEQ + n_)) * HDIM + d_] = (bf16)v;
                    else
                        ((bf16*)out)[((size_t)((b_ * HEADS + h_) * HDIM + d_)) * SEQ + n_] = (bf16)v;
                }
            }
        }
    }
}

// ---------------------------------------------------------------------------
// Flash attention: one wave per (b, h, 16-query tile).
// Q pre-scaled by log2(e)/sqrt(D) so softmax uses exp2 directly.
// Q,K: [B,H,SEQ,D] bf16;  V: [B,H,D,SEQ] bf16 (transposed);  O: [B,SEQ,EMB] bf16
// P staging uses interleaved columns (slot = 2*(c&15) + (c>>4)) so stores are
// packed b32 pairs and the A-frag read is one contiguous 32B window + perm.
// ---------------------------------------------------------------------------
__global__ __launch_bounds__(256) void flash_attn(
    const bf16* __restrict__ Q, const bf16* __restrict__ Km,
    const bf16* __restrict__ Vt, bf16* __restrict__ O) {
    __shared__ __align__(64) bf16 lds[8][16 * 32];   // per-wave P tile

    const int wid = threadIdx.x >> 5;
    const int wave = blockIdx.x * 8 + wid;
    const int lane = threadIdx.x & 31;
    const int g = lane >> 4, l = lane & 15;
    const int qt = wave & (SEQ / 16 - 1);
    const int h = (wave >> 7) & (HEADS - 1);
    const int b = wave >> 11;
    const int q0 = qt << 4;

    // Preload Q A-fragments for both 32-wide halves of D=64.
    const bf16* qbase = Q + (((size_t)(b * HEADS + h)) * SEQ + q0 + l) * HDIM;
    v16bf qa[2];
#pragma unroll
    for (int kk = 0; kk < 2; ++kk) {
        const bf16* p = qbase + kk * 32 + g * 16;
        v8bf lo = *(const v8bf*)p;
        v8bf hi = *(const v8bf*)(p + 8);
        qa[kk] = CONCAT16(lo, hi);
    }

    const bf16* kbase = Km + ((size_t)(b * HEADS + h)) * SEQ * HDIM;
    const bf16* vbase = Vt + ((size_t)(b * HEADS + h)) * HDIM * SEQ;

    float mrow[8], lrow[8];
#pragma unroll
    for (int r = 0; r < 8; ++r) { mrow[r] = -3.0e38f; lrow[r] = 0.0f; }
    v8f acc[4] = {};
    bf16* myl = lds[wid];

    for (int kb = 0; kb < SEQ; kb += 32) {
        // ---- S = Q * K^T for 32 keys (two 16x16 C tiles) ----
        v8f s0 = {}, s1 = {};
        {
            const bf16* kp0 = kbase + (size_t)(kb + l) * HDIM + g * 16;
            const bf16* kp1 = kbase + (size_t)(kb + 16 + l) * HDIM + g * 16;
            v16bf b00 = *(const v16bf*)(kp0);
            v16bf b01 = *(const v16bf*)(kp0 + 32);
            v16bf b10 = *(const v16bf*)(kp1);
            v16bf b11 = *(const v16bf*)(kp1 + 32);
            s0 = wmma_bf16(qa[0], b00, s0);
            s0 = wmma_bf16(qa[1], b01, s0);
            s1 = wmma_bf16(qa[0], b10, s1);
            s1 = wmma_bf16(qa[1], b11, s1);
        }

        // ---- online softmax (C layout: VGPR r -> row r+8g, col = lane&15) ----
        float pr0[8], pr1[8];
#pragma unroll
        for (int r = 0; r < 8; ++r) {
            float v = fmaxf(s0[r], s1[r]);
            RED16_MAX(v);
            const float mn = fmaxf(mrow[r], v);
            const float corr = exp2f(mrow[r] - mn);
            mrow[r] = mn;
            const float p0 = exp2f(s0[r] - mn);
            const float p1 = exp2f(s1[r] - mn);
            pr0[r] = p0; pr1[r] = p1;
            float rs = p0 + p1;
            RED16_ADD(rs);
            lrow[r] = lrow[r] * corr + rs;
#pragma unroll
            for (int j = 0; j < 4; ++j) acc[j][r] *= corr;
        }

        // ---- C-layout -> A-layout shuffle of P through LDS (packed b32) ----
#pragma unroll
        for (int r = 0; r < 8; ++r) {
            v2bf t;
            t[0] = (bf16)pr0[r];               // col l      -> slot 2l
            t[1] = (bf16)pr1[r];               // col l + 16 -> slot 2l + 1
            *(v2bf*)&myl[(r + 8 * g) * 32 + 2 * l] = t;
        }
        asm volatile("s_wait_dscnt 0" ::: "memory");   // DS in-order per wave
        v16bf w = *(const v16bf*)&myl[l * 32 + g * 16];
        v16bf pa = DEINTERLEAVE16(w);

        // ---- O += P * V (V transposed: rows of fixed d are key-contiguous) ----
#pragma unroll
        for (int j = 0; j < 4; ++j) {
            const bf16* vp = vbase + (size_t)(j * 16 + l) * SEQ + kb + g * 8;
            v16bf vb = CONCAT16(*(const v8bf*)vp, *(const v8bf*)(vp + 16));
            acc[j] = wmma_bf16(pa, vb, acc[j]);
        }
    }

    // ---- epilogue: normalize and store bf16 [B,SEQ,EMB] ----
#pragma unroll
    for (int r = 0; r < 8; ++r) {
        const float inv = 1.0f / lrow[r];
        const int q = q0 + r + 8 * g;
        const size_t rowbase = ((size_t)(b * SEQ + q)) * EMB + h * HDIM;
#pragma unroll
        for (int j = 0; j < 4; ++j)
            O[rowbase + j * 16 + l] = (bf16)(acc[j][r] * inv);
    }
}

// ---------------------------------------------------------------------------
extern "C" void kernel_launch(void* const* d_in, const int* in_sizes, int n_in,
                              void* d_out, int out_size, void* d_ws, size_t ws_size,
                              hipStream_t stream) {
    const float* x_q = (const float*)d_in[0];
    const float* x_k = (const float*)d_in[1];
    const float* x_v = (const float*)d_in[2];
    const float* Wq = (const float*)d_in[3];
    const float* bq = (const float*)d_in[4];
    const float* Wk = (const float*)d_in[5];
    const float* bk = (const float*)d_in[6];
    const float* Wv = (const float*)d_in[7];
    const float* bv = (const float*)d_in[8];
    const float* Wo = (const float*)d_in[9];
    const float* bo = (const float*)d_in[10];

    const size_t MK = (size_t)MTOT * EMB;   // 8M elems
    const size_t WK = (size_t)EMB * EMB;    // 1M elems

    bf16* p = (bf16*)d_ws;
    bf16* xq = p; p += MK;
    bf16* xk = p; p += MK;
    bf16* xv = p; p += MK;
    bf16* wq = p; p += WK;
    bf16* wk = p; p += WK;
    bf16* wv = p; p += WK;
    bf16* wo = p; p += WK;
    bf16* Qb = p; p += MK;   // [B,H,SEQ,D], pre-scaled
    bf16* Kb = p; p += MK;   // [B,H,SEQ,D]
    bf16* Vb = p; p += MK;   // [B,H,D,SEQ]
    bf16* AO = p; p += MK;   // attention out [B,SEQ,EMB]

    const int cvtMblk = (int)(MK / 4 / 256);   // 8192
    const int cvtWblk = (int)(WK / 4 / 256);   // 1024
    cvt_f32_to_bf16<<<cvtMblk, 256, 0, stream>>>(x_q, xq, (int)(MK / 4));
    cvt_f32_to_bf16<<<cvtMblk, 256, 0, stream>>>(x_k, xk, (int)(MK / 4));
    cvt_f32_to_bf16<<<cvtMblk, 256, 0, stream>>>(x_v, xv, (int)(MK / 4));
    cvt_f32_to_bf16<<<cvtWblk, 256, 0, stream>>>(Wq, wq, (int)(WK / 4));
    cvt_f32_to_bf16<<<cvtWblk, 256, 0, stream>>>(Wk, wk, (int)(WK / 4));
    cvt_f32_to_bf16<<<cvtWblk, 256, 0, stream>>>(Wv, wv, (int)(WK / 4));
    cvt_f32_to_bf16<<<cvtWblk, 256, 0, stream>>>(Wo, wo, (int)(WK / 4));

    // blocks = (8192/256) * (1024/64) = 32 * 16 = 512, 8 waves each
    const int gblk = (MTOT / 256) * (EMB / 64);
    const float qscale = 0.18033688011112043f;  // log2(e) / sqrt(64)
    gemm_xWT<1><<<gblk, 256, 0, stream>>>(xq, wq, bq, Qb, MTOT, EMB, EMB, qscale);
    gemm_xWT<1><<<gblk, 256, 0, stream>>>(xk, wk, bk, Kb, MTOT, EMB, EMB, 1.0f);
    gemm_xWT<2><<<gblk, 256, 0, stream>>>(xv, wv, bv, Vb, MTOT, EMB, EMB, 1.0f);

    // attention: B*H*(SEQ/16) = 8192 waves -> 1024 blocks
    flash_attn<<<BATCH * HEADS * (SEQ / 16) / 8, 256, 0, stream>>>(Qb, Kb, Vb, AO);

    gemm_xWT<0><<<gblk, 256, 0, stream>>>(AO, wo, bo, d_out, MTOT, EMB, EMB, 1.0f);
}
